// Model_36532991819832
// MI455X (gfx1250) — matrix operand order repaired and checked by reading, not compiled, not run on hardware
//
#include <hip/hip_runtime.h>
#include <hip/hip_bf16.h>

typedef __attribute__((ext_vector_type(16))) __bf16 v16bf;
typedef __attribute__((ext_vector_type(8)))  __bf16 v8bf;
typedef __attribute__((ext_vector_type(8)))  float  v8f;

#define B_    16384
#define L_    20
#define F_    28
#define E_    20
#define H_    188
#define HP    192
#define HID_  752
#define HIDP  768
#define OUT_  1000
#define OUTP  1024
#define NBLK_ 10
#define ALPHA_ 0.9f

__constant__ int d_seg[13] = {0,20,26,47,67,73,94,114,120,141,161,167,188};

// ---------------------------------------------------------------------------
// Kernel 1: pooling + SE prep. One 192-thread block per sample.
// Writes h padded to 192 cols: fp32 (residual accumulator) + bf16 (WMMA A).
// ---------------------------------------------------------------------------
__global__ __launch_bounds__(192) void prep_kernel(
    const float* __restrict__ x, const int* __restrict__ mask,
    const float* __restrict__ embed,
    const float* __restrict__ se_w1, const float* __restrict__ se_b1,
    const float* __restrict__ se_w2, const float* __restrict__ se_b2,
    float* __restrict__ h32, __bf16* __restrict__ hbf)
{
    int b = blockIdx.x;
    int d = threadIdx.x;
    __shared__ float sh[H_];
    __shared__ float ssm[12];
    __shared__ float st1[3];
    __shared__ float sse[12];

    int len = 0;
    #pragma unroll
    for (int l = 0; l < L_; ++l) len += mask[b * L_ + l];

    const float* xb = x + (size_t)b * L_ * F_;

    float hval = 0.0f;
    if (d < H_) {
        int c = d / 47, j = d % 47;
        auto fetch = [&](int l) -> float {
            if (j < E_) {
                int id = (int)xb[l * F_ + 0];
                return embed[(size_t)id * E_ + j];
            }
            return xb[l * F_ + (j - E_ + 1)];
        };
        if (c == 0) {                      // time-decay weighted sum
            float wsum = (1.0f - powf(ALPHA_, (float)len)) / (1.0f - ALPHA_);
            float td = 0.0f;
            for (int l = 0; l < len; ++l)
                td += powf(ALPHA_, (float)(len - 1 - l)) * fetch(l);
            hval = td / wsum;
        } else if (c == 1) {               // masked max, clipped
            float mx = -3.4e38f;
            for (int l = 0; l < len; ++l) mx = fmaxf(mx, fetch(l));
            hval = fminf(fmaxf(mx, 1e-9f), 1e9f);
        } else if (c == 2) {               // masked min, clipped
            float mn = 3.4e38f;
            for (int l = 0; l < len; ++l) mn = fminf(mn, fetch(l));
            hval = fminf(fmaxf(mn, 1e-9f), 1e9f);
        } else {                           // next item = last position
            hval = fetch(L_ - 1);
        }
        sh[d] = hval;
    }
    __syncthreads();
    if (d < 12) {                          // segment means
        float s = 0.0f;
        int a = d_seg[d], e = d_seg[d + 1];
        for (int k = a; k < e; ++k) s += sh[k];
        ssm[d] = s / (float)(e - a);
    }
    __syncthreads();
    if (d < 3) {                           // SE layer 1 + relu
        float t = se_b1[d];
        for (int i = 0; i < 12; ++i) t += ssm[i] * se_w1[i * 3 + d];
        st1[d] = fmaxf(t, 0.0f);
    }
    __syncthreads();
    if (d < 12) {                          // SE layer 2 + sigmoid
        float t = se_b2[d];
        for (int k = 0; k < 3; ++k) t += st1[k] * se_w2[k * 12 + d];
        sse[d] = 1.0f / (1.0f + expf(-t));
    }
    __syncthreads();
    float outv = 0.0f;
    if (d < H_) {
        int seg = 0;
        #pragma unroll
        for (int i = 0; i < 12; ++i) if (d >= d_seg[i]) seg = i;
        outv = hval * sse[seg];
    }
    h32[(size_t)b * HP + d] = outv;        // cols 188..191 are zero padding
    hbf[(size_t)b * HP + d] = (__bf16)outv;
}

// ---------------------------------------------------------------------------
// Kernel 2: weight transpose+pad fp32 KxN -> bf16 NxK (WMMA-B friendly:
// one column of B = one contiguous row of Wt).
// ---------------------------------------------------------------------------
__global__ void tpose_kernel(const float* __restrict__ W, __bf16* __restrict__ Wt,
                             int K, int N, int Kp, int Np)
{
    int gid = blockIdx.x * blockDim.x + threadIdx.x;
    if (gid >= Np * Kp) return;
    int n = gid / Kp, k = gid - n * Kp;
    float v = (k < K && n < N) ? W[(size_t)k * N + n] : 0.0f;
    Wt[(size_t)n * Kp + k] = (__bf16)v;
}

// ---------------------------------------------------------------------------
// Kernel 3: bf16 WMMA GEMM  D = act(A @ B + bias) [+ residual]
//   MODE 0: relu, store bf16                         (block layer 1)
//   MODE 1: relu + residual(f32), store f32 + bf16   (block layer 2)
//   MODE 2: plain, store f32 with Nreal guard        (final head)
// Workgroup tile 128M x 64N; 8 waves, wave tile 16M x 64N (4 WMMA n-tiles).
// The 64xK B tile (contiguous in global since ldb==K) is staged in LDS once
// per workgroup (cuts B VMEM traffic 8x); the MMA loop reads B fragments with
// ds_load_b128 (short waits) while A comes from global (compiler hoists).
// LDS rows padded by 8 bf16: keeps 16B alignment and makes the 16 lanes'
// starting banks disjoint 4-bank groups (conflict-free ds_load_b128).
// Grid.x = Np/64 exactly -> no predication in hot path, EXEC all-ones.
// ---------------------------------------------------------------------------
template<int MODE, int K>
__global__ __launch_bounds__(256) void gemm_wmma(
    const __bf16* __restrict__ A, int lda,
    const __bf16* __restrict__ Bt,           // NxK row-major, ldb == K
    const float* __restrict__ bias, int Nreal,
    __bf16* __restrict__ outbf, int ldo,
    float* __restrict__ fptr, int ldf)
{
    extern __shared__ __bf16 sB[];           // 64 x (K+8)
    constexpr int LSTR = K + 8;

    int tid  = threadIdx.x;
    int lane = tid & 31;
    int wid  = tid >> 5;       // 8 M subtiles of 16 rows
    int half = lane >> 4;      // 0: lanes 0-15, 1: lanes 16-31
    int l15  = lane & 15;

    int mBase = blockIdx.y * 128 + wid * 16;
    int nBase = blockIdx.x * 64;

    // ---- cooperative stage: global B tile -> LDS (vectorized 16B chunks) ----
    {
        const __bf16* src = Bt + (size_t)nBase * K;
        constexpr int CPR = K / 8;           // 16B chunks per row
        constexpr int CH  = 64 * CPR;        // total chunks (multiple of 256)
        #pragma unroll
        for (int c = tid; c < CH; c += 256) {
            int row = c / CPR;
            int off = (c - row * CPR) * 8;
            *(v8bf*)&sB[row * LSTR + off] =
                *(const v8bf*)(src + (size_t)row * K + off);
        }
    }
    __syncthreads();

    v8f acc[4];
    #pragma unroll
    for (int t = 0; t < 4; ++t) {
        int c = nBase + t * 16 + l15;
        float bv = (c < Nreal) ? bias[c] : 0.0f;   // per-lane select, no branch
        #pragma unroll
        for (int r = 0; r < 8; ++r) acc[t][r] = bv;
    }

    const __bf16* aRow  = A + (size_t)(mBase + l15) * lda + half * 16;
    const __bf16* bBase = sB + (size_t)l15 * LSTR + half * 16;

    // A fragment: lane holds 16 contiguous K of row (mBase + l15);
    // lanes 0-15 hold kk..kk+15; lanes 16-31 hold kk+16..kk+31.
    // B fragment: lane holds 16 contiguous K of column (nTile + l15);
    // lanes 0-15 start at kk, lanes 16-31 at kk+16.
    #pragma unroll
    for (int kk = 0; kk < K; kk += 32) {
        v8bf a0 = *(const v8bf*)(aRow + kk);
        v8bf a1 = *(const v8bf*)(aRow + kk + 8);
        v16bf afrag = __builtin_shufflevector(a0, a1,
            0,1,2,3,4,5,6,7,8,9,10,11,12,13,14,15);
        #pragma unroll
        for (int t = 0; t < 4; ++t) {
            const __bf16* bp = bBase + t * 16 * LSTR + kk;
            v8bf b0 = *(const v8bf*)(bp);
            v8bf b1 = *(const v8bf*)(bp + 8);
            v16bf bfrag = __builtin_shufflevector(b0, b1,
                0,1,2,3,4,5,6,7,8,9,10,11,12,13,14,15);
            acc[t] = __builtin_amdgcn_wmma_f32_16x16x32_bf16(
                false, afrag, false, bfrag, (short)0, acc[t], false, false);
        }
    }

    // Epilogue. C/D layout: VGPR r, lane l -> row r + 8*(l>=16), col l&15.
    #pragma unroll
    for (int t = 0; t < 4; ++t) {
        int ccol = nBase + t * 16 + l15;
        #pragma unroll
        for (int r = 0; r < 8; ++r) {
            int crow = mBase + r + half * 8;
            float v = acc[t][r];
            if (MODE == 0) {
                v = fmaxf(v, 0.0f);
                outbf[(size_t)crow * ldo + ccol] = (__bf16)v;
            } else if (MODE == 1) {
                v = fmaxf(v, 0.0f) + fptr[(size_t)crow * ldf + ccol];
                fptr[(size_t)crow * ldf + ccol] = v;
                outbf[(size_t)crow * ldo + ccol] = (__bf16)v;
            } else {
                if (ccol < Nreal) fptr[(size_t)crow * ldf + ccol] = v;
            }
        }
    }
}

// ---------------------------------------------------------------------------
extern "C" void kernel_launch(void* const* d_in, const int* in_sizes, int n_in,
                              void* d_out, int out_size, void* d_ws, size_t ws_size,
                              hipStream_t stream)
{
    const float* x     = (const float*)d_in[0];
    const int*   mask  = (const int*)d_in[1];
    const float* embed = (const float*)d_in[2];
    const float* se_w1 = (const float*)d_in[3];
    const float* se_b1 = (const float*)d_in[4];
    const float* se_w2 = (const float*)d_in[5];
    const float* se_b2 = (const float*)d_in[6];
    const float* rb_w1 = (const float*)d_in[7];
    const float* rb_b1 = (const float*)d_in[8];
    const float* rb_w2 = (const float*)d_in[9];
    const float* rb_b2 = (const float*)d_in[10];
    const float* fc_w  = (const float*)d_in[11];
    const float* fc_b  = (const float*)d_in[12];

    char* ws = (char*)d_ws;
    size_t off = 0;
    auto take = [&](size_t bytes) -> char* {
        char* p = ws + off;
        off = (off + bytes + 255) & ~(size_t)255;
        return p;
    };
    float*  h32 = (float*) take((size_t)B_ * HP * 4);       // fp32 residual h
    __bf16* hbf = (__bf16*)take((size_t)B_ * HP * 2);       // bf16 h (A operand)
    __bf16* z1  = (__bf16*)take((size_t)B_ * HIDP * 2);     // bf16 hidden
    __bf16* wt1 = (__bf16*)take((size_t)NBLK_ * HIDP * HP * 2);
    __bf16* wt2 = (__bf16*)take((size_t)NBLK_ * HP * HIDP * 2);
    __bf16* wtf = (__bf16*)take((size_t)OUTP * HP * 2);

    prep_kernel<<<B_, 192, 0, stream>>>(x, mask, embed, se_w1, se_b1,
                                        se_w2, se_b2, h32, hbf);

    int tpg = (HIDP * HP + 255) / 256;
    for (int i = 0; i < NBLK_; ++i)
        tpose_kernel<<<tpg, 256, 0, stream>>>(rb_w1 + (size_t)i * H_ * HID_,
                                              wt1 + (size_t)i * HIDP * HP,
                                              H_, HID_, HP, HIDP);
    for (int i = 0; i < NBLK_; ++i)
        tpose_kernel<<<tpg, 256, 0, stream>>>(rb_w2 + (size_t)i * HID_ * H_,
                                              wt2 + (size_t)i * HP * HIDP,
                                              HID_, H_, HIDP, HP);
    int tpf = (OUTP * HP + 255) / 256;
    tpose_kernel<<<tpf, 256, 0, stream>>>(fc_w, wtf, H_, OUT_, HP, OUTP);

    dim3 blk(256);
    size_t lds1 = (size_t)64 * (HP + 8) * sizeof(__bf16);    // 25.6 KB
    size_t lds2 = (size_t)64 * (HIDP + 8) * sizeof(__bf16);  // 99.3 KB
    for (int i = 0; i < NBLK_; ++i) {
        dim3 g1(HIDP / 64, B_ / 128);   // 16384 x 768 x 192
        gemm_wmma<0, HP><<<g1, blk, lds1, stream>>>(
            hbf, HP, wt1 + (size_t)i * HIDP * HP,
            rb_b1 + (size_t)i * HID_, HID_,
            z1, HIDP, nullptr, 0);
        dim3 g2(HP / 64, B_ / 128);     // 16384 x 192 x 768 + residual
        gemm_wmma<1, HIDP><<<g2, blk, lds2, stream>>>(
            z1, HIDP, wt2 + (size_t)i * HP * HIDP,
            rb_b2 + (size_t)i * H_, H_,
            hbf, HP, h32, HP);
    }
    dim3 gf(OUTP / 64, B_ / 128);       // 16384 x 1024 x 192 head -> fp32 out
    gemm_wmma<2, HP><<<gf, blk, lds1, stream>>>(
        hbf, HP, wtf, fc_b, OUT_,
        nullptr, 0, (float*)d_out, OUT_);
}
